// Conv1D_LowRankLSTM_Self_Attention_89876485636779
// MI455X (gfx1250) — matrix operand order repaired and checked by reading, not compiled, not run on hardware
//
#include <hip/hip_runtime.h>
#include <hip/hip_bf16.h>
#include <hip/hip_fp16.h>

#define B_    32
#define S_    1024
#define CIN_  16
#define HID_  128
#define RANK_ 16
#define G4_   512   // 4*HID
#define NCLS_ 18
#define NH_   2
#define HD_   64

typedef __attribute__((ext_vector_type(16))) _Float16 v16h;
typedef __attribute__((ext_vector_type(8)))  _Float16 v8h;
typedef __attribute__((ext_vector_type(8)))  float    v8f;
typedef __attribute__((ext_vector_type(4)))  unsigned int v4u;
typedef __attribute__((ext_vector_type(8)))  int      v8i;
typedef __attribute__((ext_vector_type(4)))  int      v4i;

__device__ __forceinline__ v8f wmma_f16(v16h a, v16h b, v8f c) {
  return __builtin_amdgcn_wmma_f32_16x16x32_f16(false, a, false, b, (short)0, c, false, false);
}

__device__ __forceinline__ float sigf(float x) { return 1.f / (1.f + __expf(-x)); }

// ---- WMMA fragment loaders (wave32 layouts per CDNA5 ISA 7.12.2) ----
// A 16x32 (MxK) f16: lane's 16 values are two contiguous 8-half runs -> 2x b128.
__device__ __forceinline__ v16h load_a_f16(const _Float16* src, int ld) {
  const int lane = threadIdx.x & 31, half = lane >> 4, m = lane & 15;
  const v8h lo = *(const v8h*)(src + m * ld + half * 8);
  const v8h hi = *(const v8h*)(src + m * ld + 16 + half * 8);
  return __builtin_shufflevector(lo, hi, 0, 1, 2, 3, 4, 5, 6, 7,
                                 8, 9, 10, 11, 12, 13, 14, 15);
}
// B 32x16 (KxN) f16 pre-swizzled to fragment order: 32 contiguous bytes/lane.
__device__ __forceinline__ v16h load_b_swz(const _Float16* base) {
  return *(const v16h*)(base + (threadIdx.x & 31) * 16);
}

// ---- one-time B-matrix swizzler: W(f32, KxN row-major) -> fragment order ----
__global__ __launch_bounds__(32) void bswz_kernel(const float* __restrict__ W, int N,
                                                  int Kvalid, int ntN,
                                                  _Float16* __restrict__ dst) {
  const int tile = blockIdx.x;
  const int kc = tile / ntN, nt = tile % ntN;
  const int lane = threadIdx.x & 31, half = lane >> 4, n = lane & 15;
  _Float16* d = dst + ((size_t)tile * 32 + lane) * 16;
#pragma unroll
  for (int e = 0; e < 16; ++e) {
    const int k = kc * 32 + (e < 8 ? half * 8 + e : 16 + half * 8 + (e - 8));
    d[e] = (k < Kvalid) ? (_Float16)W[(size_t)k * N + nt * 16 + n] : (_Float16)0.f;
  }
}

// ---- Tensor Data Mover: 2D tile (rows x dim0 elems, row stride in elems) -> LDS
// dsz: 0=1B,1=2B,2=4B. Bit packing per CDNA5 ISA 8.4 (D# group1).
__device__ __forceinline__ void tdm_load_2d(unsigned lds_off, const void* gsrc,
                                            unsigned dsz, unsigned dim0,
                                            unsigned rows, unsigned stride) {
  const unsigned long long ga = (unsigned long long)(uintptr_t)gsrc;
  const v4u g0 = { 1u,                                   // count=1, user descriptor
                   lds_off,                               // lds_addr
                   (unsigned)(ga & 0xffffffffu),          // global_addr[31:0]
                   (unsigned)((ga >> 32) & 0x01ffffffu) | (2u << 30) }; // [56:32]|type=2
  const v8i g1 = { (int)(dsz << 16),                     // data_size
                   (int)((dim0 & 0xffffu) << 16),        // tensor_dim0 lo16 (79:48)
                   (int)((dim0 >> 16) | ((rows & 0xffffu) << 16)), // dim0 hi | tensor_dim1 lo
                   (int)((rows >> 16) | (dim0 << 16)),   // dim1 hi | tile_dim0 (127:112)
                   (int)rows,                            // tile_dim1 (143:128), tile_dim2=0
                   (int)stride,                          // tensor_dim0_stride lo32 (191:160)
                   0, 0 };
  const v4i z4 = {0, 0, 0, 0};
#if defined(__clang_major__) && __clang_major__ >= 23
  const v8i z8 = {0, 0, 0, 0, 0, 0, 0, 0};
  __builtin_amdgcn_tensor_load_to_lds(g0, g1, z4, z4, z8, 0);
#else
  __builtin_amdgcn_tensor_load_to_lds(g0, g1, z4, z4, 0);
#endif
}

// ---------------- conv stack: depthwise(3, pad1) + pointwise ----------------
__global__ __launch_bounds__(256) void dsconv_kernel(
    const float* __restrict__ in, int cin, int cout,
    const float* __restrict__ dww, const float* __restrict__ dwb,
    const float* __restrict__ pww, const float* __restrict__ pwb,
    float* __restrict__ out, _Float16* __restrict__ out_h) {
  __shared__ float xin[66 * 64];
  __shared__ float dwv[64 * 64];
  const int b = blockIdx.y, s0 = blockIdx.x * 64, tid = threadIdx.x;
  for (int idx = tid; idx < 66 * cin; idx += 256) {
    const int r = idx / cin, ch = idx % cin;
    const int s = s0 - 1 + r;
    xin[idx] = (s >= 0 && s < S_) ? in[((size_t)b * S_ + s) * cin + ch] : 0.f;
  }
  __syncthreads();
  for (int idx = tid; idx < 64 * cin; idx += 256) {
    const int r = idx / cin, ch = idx % cin;
    float v = dwb[ch];
    v += dww[ch * 3 + 0] * xin[r * cin + ch];
    v += dww[ch * 3 + 1] * xin[(r + 1) * cin + ch];
    v += dww[ch * 3 + 2] * xin[(r + 2) * cin + ch];
    dwv[idx] = v;
  }
  __syncthreads();
  for (int idx = tid; idx < 64 * cout; idx += 256) {
    const int r = idx / cout, co = idx % cout;
    float acc = pwb[co];
    for (int ci = 0; ci < cin; ++ci) acc += pww[co * cin + ci] * dwv[r * cin + ci];
    const size_t o = ((size_t)b * S_ + s0 + r) * cout + co;
    if (out_h) out_h[o] = (_Float16)acc; else out[o] = acc;
  }
}

// ------------- gx = (x3 @ U_i) @ V_i + bias_i + bias_h  (rank-16) -----------
__global__ __launch_bounds__(32) void gx_kernel(
    const _Float16* __restrict__ x3h, const _Float16* __restrict__ Ui_swz,
    const _Float16* __restrict__ Vi_swz, const float* __restrict__ bias_i,
    const float* __restrict__ bias_h, float* __restrict__ gx) {
  __shared__ _Float16 tpad[16 * 32];                 // 16x16 t, K-padded to 32
  const int s0 = blockIdx.x * 16, b = blockIdx.y;
  const int lane = threadIdx.x & 31, half = lane >> 4, nc = lane & 15;
  const _Float16* xrow = x3h + ((size_t)b * S_ + s0) * HID_;
  // preload all fragments, then back-to-back WMMAs (no per-chunk stalls)
  v16h af[4], bf[4];
#pragma unroll
  for (int kc = 0; kc < 4; ++kc) {
    af[kc] = load_a_f16(xrow + kc * 32, HID_);
    bf[kc] = load_b_swz(Ui_swz + kc * 512);
  }
  v8f t = {};
#pragma unroll
  for (int kc = 0; kc < 4; ++kc) t = wmma_f16(af[kc], bf[kc], t);
#pragma unroll
  for (int r = 0; r < 8; ++r) {
    tpad[(r + half * 8) * 32 + nc]      = (_Float16)t[r];
    tpad[(r + half * 8) * 32 + 16 + nc] = (_Float16)0.f;
  }
  __syncthreads();
  v16h a2 = load_a_f16(tpad, 32);
  v16h bb = load_b_swz(Vi_swz);                      // double-buffered B stream
  for (int n = 0; n < 32; ++n) {
    v16h bnext = (n < 31) ? load_b_swz(Vi_swz + (n + 1) * 512) : bb;
    v8f c = {};
    c = wmma_f16(a2, bb, c);
    const int g = n * 16 + nc;
    const float bsum = bias_i[g] + bias_h[g];
#pragma unroll
    for (int r = 0; r < 8; ++r)
      gx[((size_t)b * S_ + s0 + r + half * 8) * G4_ + g] = c[r] + bsum;
    bb = bnext;
  }
}

// ------------- persistent single-workgroup LSTM scan (1024 steps) -----------
__global__ __launch_bounds__(256) void lstm_kernel(
    const float* __restrict__ gx, const _Float16* __restrict__ Uh_swz,
    const _Float16* __restrict__ Vh_swz, _Float16* __restrict__ hs) {
  __shared__ _Float16 h16[B_ * HID_];   // 8 KB state (f16 for WMMA A reads)
  __shared__ float    cst[B_ * HID_];   // 16 KB
  __shared__ _Float16 tpad[16 * 32];    // 1 KB
  __shared__ float    gl[16 * G4_];     // 32 KB: TDM destination, gates in-place
  const int tid = threadIdx.x, wave = tid >> 5;
  const int lane = tid & 31, half = lane >> 4, nc = lane & 15;
  for (int i = tid; i < B_ * HID_; i += 256) { h16[i] = (_Float16)0.f; cst[i] = 0.f; }
  __syncthreads();
  const unsigned gl_off = (unsigned)(uintptr_t)(void*)gl;   // LDS byte offset
  // B fragments are invariant over all 1024 steps: hoist into registers.
  v16h vhf[4];
#pragma unroll
  for (int nn = 0; nn < 4; ++nn) vhf[nn] = load_b_swz(Vh_swz + (wave * 4 + nn) * 512);
  v16h uhf[4];
  if (wave == 0) {
#pragma unroll
    for (int kc = 0; kc < 4; ++kc) uhf[kc] = load_b_swz(Uh_swz + kc * 512);
  }
  for (int s = 0; s < S_; ++s) {
    for (int mt = 0; mt < 2; ++mt) {                 // two 16-batch tiles
      if (wave == 0) {
        // async DMA gx[s] tile into gl, overlapped with the t WMMAs
        tdm_load_2d(gl_off, gx + ((size_t)(mt * 16) * S_ + s) * G4_,
                    /*dsz=*/2u, /*dim0=*/G4_, /*rows=*/16u, /*stride=*/(unsigned)S_ * G4_);
        v8f t = {};
#pragma unroll
        for (int kc = 0; kc < 4; ++kc)
          t = wmma_f16(load_a_f16(h16 + (mt * 16) * HID_ + kc * 32, HID_), uhf[kc], t);
#pragma unroll
        for (int r = 0; r < 8; ++r) {
          tpad[(r + half * 8) * 32 + nc]      = (_Float16)t[r];
          tpad[(r + half * 8) * 32 + 16 + nc] = (_Float16)0.f;
        }
        __builtin_amdgcn_s_wait_tensorcnt(0);
      }
      __syncthreads();                               // tpad + DMA'd gl ready
      // gates = gl(gx) + t @ V_h : 4 N-tiles per wave, B frags already in regs
      {
        v16h a2 = load_a_f16(tpad, 32);
#pragma unroll
        for (int nn = 0; nn < 4; ++nn) {
          const int n = wave * 4 + nn;
          v8f c = {};
          c = wmma_f16(a2, vhf[nn], c);
#pragma unroll
          for (int r = 0; r < 8; ++r) {
            const int idx = (r + half * 8) * G4_ + n * 16 + nc;
            gl[idx] = c[r] + gl[idx];
          }
        }
      }
      __syncthreads();
      // elementwise LSTM cell update for this 16-batch tile
      for (int e = tid; e < 16 * HID_; e += 256) {
        const int rl = e >> 7, j = e & 127;
        const int bglob = mt * 16 + rl;
        const float gi = gl[rl * G4_ + j];
        const float gf = gl[rl * G4_ + 128 + j];
        const float gc = gl[rl * G4_ + 256 + j];
        const float go = gl[rl * G4_ + 384 + j];
        float cc = cst[bglob * HID_ + j];
        cc = sigf(gf) * cc + sigf(gi) * tanhf(gc);
        const float hh = sigf(go) * tanhf(cc);
        cst[bglob * HID_ + j] = cc;
        h16[bglob * HID_ + j] = (_Float16)hh;
        hs[((size_t)bglob * S_ + s) * HID_ + j] = (_Float16)hh;
      }
      __syncthreads();
    }
  }
}

// ------------- K/V projections: (B*S x 128) @ (128 x 128) WMMA GEMM ---------
// A tile staged once per block via TDM; B fragments preloaded per wave.
__global__ __launch_bounds__(256) void kv_kernel(
    const _Float16* __restrict__ hs,
    const _Float16* __restrict__ Wk_swz, const float* __restrict__ bk,
    const _Float16* __restrict__ Wv_swz, const float* __restrict__ bv,
    float* __restrict__ Kb, float* __restrict__ Vb) {
  __shared__ _Float16 atile[16 * HID_];        // 4 KB hs tile
  const int s0 = blockIdx.x * 16, b = blockIdx.y, z = blockIdx.z;
  const _Float16* W = z ? Wv_swz : Wk_swz;
  const float* bias = z ? bv : bk;
  float* out        = z ? Vb : Kb;
  const int wave = threadIdx.x >> 5;           // N-tile 0..7
  const int lane = threadIdx.x & 31, half = lane >> 4, nc = lane & 15;
  if (wave == 0)
    tdm_load_2d((unsigned)(uintptr_t)(void*)atile, hs + ((size_t)b * S_ + s0) * HID_,
                /*dsz=*/1u, /*dim0=*/(unsigned)HID_, /*rows=*/16u, /*stride=*/(unsigned)HID_);
  v16h bf[4];
#pragma unroll
  for (int kc = 0; kc < 4; ++kc)
    bf[kc] = load_b_swz(W + ((size_t)kc * 8 + wave) * 512);
  if (wave == 0) __builtin_amdgcn_s_wait_tensorcnt(0);
  __syncthreads();
  v8f c = {};
#pragma unroll
  for (int kc = 0; kc < 4; ++kc)
    c = wmma_f16(load_a_f16(atile + kc * 32, HID_), bf[kc], c);
  const int col = wave * 16 + nc;
  const float bb0 = bias[col];
#pragma unroll
  for (int r = 0; r < 8; ++r)
    out[((size_t)b * S_ + s0 + r + half * 8) * HID_ + col] = c[r] + bb0;
}

// ------- last-row attention: q = hs[-1]@Wq, softmax(q.K/8).V per head -------
__global__ __launch_bounds__(256) void attn_kernel(
    const _Float16* __restrict__ hs, const float* __restrict__ Wq,
    const float* __restrict__ bq, const float* __restrict__ Kb,
    const float* __restrict__ Vb, float* __restrict__ ao) {
  __shared__ float q[HD_];
  __shared__ float sc[S_];
  __shared__ float red[256];
  __shared__ float part[4 * HD_];
  const int b = blockIdx.x, h = blockIdx.y, tid = threadIdx.x;
  if (tid < HD_) {
    float acc = bq[h * HD_ + tid];
    for (int j = 0; j < HID_; ++j)
      acc += (float)hs[((size_t)b * S_ + (S_ - 1)) * HID_ + j] * Wq[j * HID_ + h * HD_ + tid];
    q[tid] = acc;
  }
  __syncthreads();
  float lmax = -1e30f;
  for (int k = tid; k < S_; k += 256) {
    float s = 0.f;
    for (int d = 0; d < HD_; ++d)
      s += q[d] * Kb[((size_t)b * S_ + k) * HID_ + h * HD_ + d];
    s *= 0.125f;
    sc[k] = s;
    lmax = fmaxf(lmax, s);
  }
  red[tid] = lmax;
  __syncthreads();
  for (int st = 128; st > 0; st >>= 1) {
    if (tid < st) red[tid] = fmaxf(red[tid], red[tid + st]);
    __syncthreads();
  }
  const float mx = red[0];
  __syncthreads();
  float lsum = 0.f;
  for (int k = tid; k < S_; k += 256) {
    const float e = __expf(sc[k] - mx);
    sc[k] = e;
    lsum += e;
  }
  red[tid] = lsum;
  __syncthreads();
  for (int st = 128; st > 0; st >>= 1) {
    if (tid < st) red[tid] += red[tid + st];
    __syncthreads();
  }
  const float inv = 1.f / red[0];
  __syncthreads();
  const int d = tid & 63, kg = tid >> 6;
  float acc = 0.f;
  for (int k = kg * 256; k < kg * 256 + 256; ++k)
    acc += sc[k] * Vb[((size_t)b * S_ + k) * HID_ + h * HD_ + d];
  part[kg * HD_ + d] = acc;
  __syncthreads();
  if (tid < HD_)
    ao[b * HID_ + h * HD_ + tid] =
        (part[tid] + part[HD_ + tid] + part[2 * HD_ + tid] + part[3 * HD_ + tid]) * inv;
}

// ---------------- final: (ao @ Wo + bo) @ Wfc + bfc -> (32,18) --------------
__global__ __launch_bounds__(256) void final_kernel(
    const float* __restrict__ ao, const float* __restrict__ Wo,
    const float* __restrict__ bo, const float* __restrict__ Wfc,
    const float* __restrict__ bfc, float* __restrict__ out) {
  __shared__ float o1[B_ * HID_];
  const int tid = threadIdx.x;
  for (int idx = tid; idx < B_ * HID_; idx += 256) {
    const int bb = idx >> 7, j = idx & 127;
    float acc = bo[j];
    for (int d = 0; d < HID_; ++d) acc += ao[bb * HID_ + d] * Wo[d * HID_ + j];
    o1[idx] = acc;
  }
  __syncthreads();
  for (int idx = tid; idx < B_ * NCLS_; idx += 256) {
    const int bb = idx / NCLS_, n = idx % NCLS_;
    float acc = bfc[n];
    for (int d = 0; d < HID_; ++d) acc += o1[bb * HID_ + d] * Wfc[d * NCLS_ + n];
    out[idx] = acc;
  }
}

extern "C" void kernel_launch(void* const* d_in, const int* in_sizes, int n_in,
                              void* d_out, int out_size, void* d_ws, size_t ws_size,
                              hipStream_t stream) {
  const float* x     = (const float*)d_in[0];
  const float* c1dww = (const float*)d_in[1];
  const float* c1dwb = (const float*)d_in[2];
  const float* c1pww = (const float*)d_in[3];
  const float* c1pwb = (const float*)d_in[4];
  const float* c2dww = (const float*)d_in[5];
  const float* c2dwb = (const float*)d_in[6];
  const float* c2pww = (const float*)d_in[7];
  const float* c2pwb = (const float*)d_in[8];
  const float* c3dww = (const float*)d_in[9];
  const float* c3dwb = (const float*)d_in[10];
  const float* c3pww = (const float*)d_in[11];
  const float* c3pwb = (const float*)d_in[12];
  const float* U_i   = (const float*)d_in[13];
  const float* V_i   = (const float*)d_in[14];
  const float* U_h   = (const float*)d_in[15];
  const float* V_h   = (const float*)d_in[16];
  const float* bias_i = (const float*)d_in[17];
  const float* bias_h = (const float*)d_in[18];
  const float* Wq = (const float*)d_in[19];
  const float* bq = (const float*)d_in[20];
  const float* Wk = (const float*)d_in[21];
  const float* bk = (const float*)d_in[22];
  const float* Wv = (const float*)d_in[23];
  const float* bv = (const float*)d_in[24];
  const float* Wo = (const float*)d_in[25];
  const float* bo = (const float*)d_in[26];
  const float* Wfc = (const float*)d_in[27];
  const float* bfc = (const float*)d_in[28];
  float* out = (float*)d_out;

  char* ws = (char*)d_ws;
  float*    x1   = (float*)(ws + 0);                      // B*S*32  f32 :  4 MB
  float*    x2   = (float*)(ws + (size_t)4194304);        // B*S*64  f32 :  8 MB
  _Float16* x3h  = (_Float16*)(ws + (size_t)12582912);    // B*S*128 f16 :  8 MB
  float*    gx   = (float*)(ws + (size_t)20971520);       // B*S*512 f32 : 64 MB
  _Float16* hsb  = (_Float16*)(ws + (size_t)88080384);    // B*S*128 f16 :  8 MB
  float*    Kb   = (float*)(ws + (size_t)96468992);       // B*S*128 f32 : 16 MB
  float*    Vb   = (float*)(ws + (size_t)113246208);      // B*S*128 f32 : 16 MB
  float*    ao   = (float*)(ws + (size_t)130023424);      // B*128   f32
  _Float16* Ui_swz = (_Float16*)(ws + (size_t)130039808); //  4 KB (4 tiles)
  _Float16* Uh_swz = (_Float16*)(ws + (size_t)130043904); //  4 KB
  _Float16* Vi_swz = (_Float16*)(ws + (size_t)130048000); // 32 KB (32 tiles, K16-pad)
  _Float16* Vh_swz = (_Float16*)(ws + (size_t)130080768); // 32 KB
  _Float16* Wk_swz = (_Float16*)(ws + (size_t)130113536); // 32 KB (4x8 tiles)
  _Float16* Wv_swz = (_Float16*)(ws + (size_t)130146304); // 32 KB

  bswz_kernel<<<dim3(4),  32, 0, stream>>>(U_i, RANK_, HID_, 1,  Ui_swz);
  bswz_kernel<<<dim3(4),  32, 0, stream>>>(U_h, RANK_, HID_, 1,  Uh_swz);
  bswz_kernel<<<dim3(32), 32, 0, stream>>>(V_i, G4_,  RANK_, 32, Vi_swz);
  bswz_kernel<<<dim3(32), 32, 0, stream>>>(V_h, G4_,  RANK_, 32, Vh_swz);
  bswz_kernel<<<dim3(32), 32, 0, stream>>>(Wk,  HID_, HID_,  8,  Wk_swz);
  bswz_kernel<<<dim3(32), 32, 0, stream>>>(Wv,  HID_, HID_,  8,  Wv_swz);

  dsconv_kernel<<<dim3(16, B_), 256, 0, stream>>>(x,  16, 32,  c1dww, c1dwb, c1pww, c1pwb, x1, nullptr);
  dsconv_kernel<<<dim3(16, B_), 256, 0, stream>>>(x1, 32, 64,  c2dww, c2dwb, c2pww, c2pwb, x2, nullptr);
  dsconv_kernel<<<dim3(16, B_), 256, 0, stream>>>(x2, 64, 128, c3dww, c3dwb, c3pww, c3pwb, nullptr, x3h);
  gx_kernel<<<dim3(S_ / 16, B_), 32, 0, stream>>>(x3h, Ui_swz, Vi_swz, bias_i, bias_h, gx);
  lstm_kernel<<<1, 256, 0, stream>>>(gx, Uh_swz, Vh_swz, hsb);
  kv_kernel<<<dim3(S_ / 16, B_, 2), 256, 0, stream>>>(hsb, Wk_swz, bk, Wv_swz, bv, Kb, Vb);
  attn_kernel<<<dim3(B_, NH_), 256, 0, stream>>>(hsb, Wq, bq, Kb, Vb, ao);
  final_kernel<<<1, 256, 0, stream>>>(ao, Wo, bo, Wfc, bfc, out);
}
